// SelfAttention_torch_79654463472367
// MI455X (gfx1250) — compile-verified, matrix operand order checked by reading, not run on hardware
//
#include <hip/hip_runtime.h>

// ---------------------------------------------------------------------------
// Self-attention forward for MI455X (gfx1250, wave32, WMMA bf16 16x16x32).
//   B=4, S=2048, H=16, DH=64, D=1024.
// Phase 0: cvt kernels        : f32 -> bf16 for activations and weights
// Phase 1: qkv_proj_kernel x3 : y = (x @ W^T + b) * outScale  (bf16 in/out)
//          wave tile 32m x 64n; Q pre-scaled by 1/sqrt(DH); V stored transposed
// Phase 2: attn_kernel        : flash attention, 128 q rows/block. K & V^T
//          tiles are DMA'd into double-buffered LDS with
//          global_load_async_to_lds_b128 (ASYNCcnt), prefetching tile kt+1
//          while tile kt is computed. Online softmax; row sums via WMMA
//          against a ones matrix; fp32 out.
// ---------------------------------------------------------------------------

typedef __attribute__((ext_vector_type(16))) __bf16 bf16x16;
typedef __attribute__((ext_vector_type(8)))  float  f32x8;

struct Frag16 {
    union {
        bf16x16  v;
        unsigned u[8];
    };
};

// fast f32 -> bf16 (round-half-up): bias add then take high half.
__device__ __forceinline__ unsigned short f2bf(float f) {
    return (unsigned short)((__float_as_uint(f) + 0x8000u) >> 16);
}
// pack two: one v_perm_b32 selecting the high halves of the rounded values.
__device__ __forceinline__ unsigned pk2(float lo, float hi) {
    unsigned a = __float_as_uint(lo) + 0x8000u;
    unsigned b = __float_as_uint(hi) + 0x8000u;
    return __builtin_amdgcn_perm(b, a, 0x07060302u);
}

// bf16 operand fragment: two 16-byte loads. p points at element K=kb of the
// row; chunks are K=kb..kb+7 and K=kb+16..kb+23, matching the CDNA5 16-bit
// A/B operand VGPR layout (lane kb = (lane>>4)*8).
__device__ __forceinline__ void load_frag_bf16(Frag16& d, const unsigned short* __restrict__ p) {
    *(uint4*)(&d.u[0]) = *(const uint4*)(p);
    *(uint4*)(&d.u[4]) = *(const uint4*)(p + 16);
}

__device__ __forceinline__ f32x8 bwmma(const Frag16& a, const Frag16& b, f32x8 c) {
    return __builtin_amdgcn_wmma_f32_16x16x32_bf16(
        /*neg_a=*/false, a.v, /*neg_b=*/false, b.v,
        /*c_mod=*/(short)0, c, /*reuse_a=*/false, /*reuse_b=*/false);
}

__device__ __forceinline__ f32x8 zero8() {
    f32x8 z = {0.f, 0.f, 0.f, 0.f, 0.f, 0.f, 0.f, 0.f};
    return z;
}

// Async DMA of 16 bytes global -> LDS (per lane), tracked by ASYNCcnt.
// Inline asm because the clang builtin arity differs across toolchains.
// LDS byte address = low 32 bits of the generic pointer (flat-LDS mapping).
__device__ __forceinline__ void async_copy_b128(void* lds_dst, const void* gsrc) {
    unsigned dst = (unsigned)(unsigned long long)lds_dst;
    unsigned long long src = (unsigned long long)gsrc;
    asm volatile("global_load_async_to_lds_b128 %0, %1, off"
                 :: "v"(dst), "v"(src) : "memory");
}
__device__ __forceinline__ void wait_async0() {
    asm volatile("s_wait_asynccnt 0x0" ::: "memory");
}

// max-reduction across each 16-lane half of the wave32 (C/D layout: a matrix
// row lives in one VGPR slot across lanes 0-15 or 16-31). xor swizzles.
__device__ __forceinline__ float swzmax16(float x) {
    x = fmaxf(x, __int_as_float(__builtin_amdgcn_ds_swizzle(__float_as_int(x), 0x041F)));
    x = fmaxf(x, __int_as_float(__builtin_amdgcn_ds_swizzle(__float_as_int(x), 0x081F)));
    x = fmaxf(x, __int_as_float(__builtin_amdgcn_ds_swizzle(__float_as_int(x), 0x101F)));
    x = fmaxf(x, __int_as_float(__builtin_amdgcn_ds_swizzle(__float_as_int(x), 0x201F)));
    return x;
}

// ---------------------------------------------------------------------------
// Phase 0: elementwise f32 -> bf16, 8 elements per thread (b128 in/out).
// ---------------------------------------------------------------------------
__global__ __launch_bounds__(256) void cvt_bf16_kernel(
    const float* __restrict__ src, unsigned short* __restrict__ dst)
{
    const size_t base = ((size_t)blockIdx.x * 256 + threadIdx.x) * 8;
    float4 a = *(const float4*)(src + base);
    float4 b = *(const float4*)(src + base + 4);
    uint4 o;
    o.x = pk2(a.x, a.y);
    o.y = pk2(a.z, a.w);
    o.z = pk2(b.x, b.y);
    o.w = pk2(b.z, b.w);
    *(uint4*)(dst + base) = o;
}

// ---------------------------------------------------------------------------
// Phase 1: projection GEMM (all-bf16 operands). grid = (M/32, N/512),
// block = 256 (8 waves). Wave: 32(m) x 64(n) tile, 8 accumulators, K loop
// 1024 step 32; each B fragment feeds two WMMA (24 bytes loaded per WMMA).
// storeVT==0 -> out[((b*16+h)*2048+s)*64+dh]   (bf16, head-major)
// storeVT==1 -> out[((b*16+h)*64+dh)*2048+s]   (bf16, transposed V)
// ---------------------------------------------------------------------------
__global__ __launch_bounds__(256) void qkv_proj_kernel(
    const unsigned short* __restrict__ X, const unsigned short* __restrict__ W,
    const float* __restrict__ bias, unsigned short* __restrict__ out,
    int storeVT, float outScale)
{
    const int lane  = threadIdx.x & 31;
    const int wv    = threadIdx.x >> 5;
    const int lo16  = lane & 15;
    const int hi    = lane >> 4;
    const int kb    = hi * 8;
    const int mbase = blockIdx.x * 32;
    const int nbase = blockIdx.y * 512 + wv * 64;

    const unsigned short* __restrict__ xrow0 = X + (size_t)(mbase + lo16) * 1024;
    const unsigned short* __restrict__ xrow1 = X + (size_t)(mbase + 16 + lo16) * 1024;

    f32x8 acc[8];
#pragma unroll
    for (int i = 0; i < 8; ++i) acc[i] = zero8();

#pragma unroll 1
    for (int kk = 0; kk < 1024; kk += 32) {
        Frag16 a0, a1;
        load_frag_bf16(a0, xrow0 + kk + kb);
        load_frag_bf16(a1, xrow1 + kk + kb);
#pragma unroll
        for (int t = 0; t < 4; ++t) {
            // B operand lane n reads row n of W (y = x @ W^T), contiguous.
            const unsigned short* wrow = W + (size_t)(nbase + 16 * t + lo16) * 1024 + kk + kb;
            Frag16 b;
            load_frag_bf16(b, wrow);
            acc[t]     = bwmma(a0, b, acc[t]);
            acc[4 + t] = bwmma(a1, b, acc[4 + t]);
        }
    }

#pragma unroll
    for (int g = 0; g < 2; ++g) {
#pragma unroll
        for (int t = 0; t < 4; ++t) {
            const int n  = nbase + 16 * t + lo16;
            const float bv = bias[n];
            const int hh = n >> 6;
            const int dh = n & 63;
#pragma unroll
            for (int r = 0; r < 8; ++r) {
                const int m  = mbase + 16 * g + r + 8 * hi;  // global row in [0, B*S)
                const int bb = m >> 11;                      // m / 2048
                const int ss = m & 2047;                     // m % 2048
                const float val = (acc[4 * g + t][r] + bv) * outScale;
                const size_t idx = storeVT
                    ? ((size_t)((bb * 16 + hh) * 64 + dh) * 2048 + (size_t)ss)
                    : ((size_t)((bb * 16 + hh) * 2048 + ss) * 64 + (size_t)dh);
                out[idx] = f2bf(val);
            }
        }
    }
}

// ---------------------------------------------------------------------------
// Phase 2: flash attention. grid = B*H*(S/128) = 1024 blocks, 256 threads
// (8 waves, one 16-row q strip each; all share the block's K/V LDS tiles).
// Double-buffered LDS tiles filled by async-to-LDS DMA, prefetched one tile
// ahead of compute. LDS row strides padded (72 / 40 ushorts) so consecutive
// rows land in distinct bank groups. Q pre-scaled by 1/sqrt(DH).
// ---------------------------------------------------------------------------
__global__ __launch_bounds__(256) void attn_kernel(
    const unsigned short* __restrict__ Qp,   // [b,h,s,dh] bf16 (scaled)
    const unsigned short* __restrict__ Kp,   // [b,h,s,dh] bf16
    const unsigned short* __restrict__ Vt,   // [b,h,dh,s] bf16
    float* __restrict__ out)                 // [b,s,h*64+dh] f32
{
    __shared__ unsigned short kbuf[2][32 * 72];    // 32 keys x 64 dh (stride 72)
    __shared__ unsigned short vbuf[2][64 * 40];    // 64 dh x 32 keys (stride 40)
    __shared__ unsigned short pbuf[8 * 16 * 40];   // per-wave P (stride 40)

    const int tid   = threadIdx.x;
    const int lane  = tid & 31;
    const int wv    = tid >> 5;
    const int lo16  = lane & 15;
    const int hi    = lane >> 4;
    const int kb    = hi * 8;

    const int qblk  = blockIdx.x & 15;       // S/128 = 16
    const int bh    = blockIdx.x >> 4;       // b*16 + h
    const int bb    = bh >> 4;
    const int hh    = bh & 15;
    const int qbase = qblk * 128 + wv * 16;

    // Q fragments (dh chunks 0..31, 32..63) stay in registers for all of Sk.
    Frag16 qf[2];
    {
        const unsigned short* qptr = Qp + (size_t)(bh * 2048 + qbase + lo16) * 64;
        load_frag_bf16(qf[0], qptr + kb);
        load_frag_bf16(qf[1], qptr + 32 + kb);
    }

    // constant ones matrix (bf16 1.0 = 0x3F80) for WMMA row sums
    Frag16 ones;
#pragma unroll
    for (int i = 0; i < 8; ++i) ones.u[i] = 0x3F803F80u;

    const unsigned short* __restrict__ kbaseP = Kp + (size_t)bh * 2048 * 64;
    const unsigned short* __restrict__ vbaseP = Vt + (size_t)bh * 64 * 2048;
    unsigned short* myp = pbuf + wv * (16 * 40);

    // staging indices: K tile = 32 rows x 64; V^T tile = 64 rows x 32
    const int krow = tid >> 3, kcol = (tid & 7) * 8;
    const int vrow = tid >> 2, vcol = (tid & 3) * 8;

    // prologue: DMA tile 0 into buffer 0
    async_copy_b128(&kbuf[0][krow * 72 + kcol],
                    kbaseP + (size_t)krow * 64 + kcol);
    async_copy_b128(&vbuf[0][vrow * 40 + vcol],
                    vbaseP + (size_t)vrow * 2048 + vcol);
    wait_async0();
    __syncthreads();

    f32x8 o[4];
    o[0] = zero8(); o[1] = zero8(); o[2] = zero8(); o[3] = zero8();
    float mrow[8], lrow[8];
#pragma unroll
    for (int r = 0; r < 8; ++r) { mrow[r] = -3.0e38f; lrow[r] = 0.0f; }

#pragma unroll 1
    for (int kt = 0; kt < 64; ++kt) {
        const int cur = kt & 1;

        // ---- prefetch tile kt+1 into the other buffer (overlaps compute) ----
        if (kt < 63) {
            const int kgn = (kt + 1) * 32;
            async_copy_b128(&kbuf[cur ^ 1][krow * 72 + kcol],
                            kbaseP + (size_t)(kgn + krow) * 64 + kcol);
            async_copy_b128(&vbuf[cur ^ 1][vrow * 40 + vcol],
                            vbaseP + (size_t)vrow * 2048 + kgn + vcol);
        }

        // ---- S = Q K^T : two 16x16 score tiles, each 2 WMMA over dh ----
        f32x8 s0 = zero8(), s1 = zero8();
        {
            const unsigned short* kp0 = &kbuf[cur][lo16 * 72];
            const unsigned short* kp1 = &kbuf[cur][(16 + lo16) * 72];
            Frag16 kf;
            load_frag_bf16(kf, kp0 + kb);        s0 = bwmma(qf[0], kf, s0);
            load_frag_bf16(kf, kp0 + 32 + kb);   s0 = bwmma(qf[1], kf, s0);
            load_frag_bf16(kf, kp1 + kb);        s1 = bwmma(qf[0], kf, s1);
            load_frag_bf16(kf, kp1 + 32 + kb);   s1 = bwmma(qf[1], kf, s1);
        }

        // ---- online softmax: row max via swizzles, exponentials ----
        float alpha[8];
#pragma unroll
        for (int r = 0; r < 8; ++r) {
            const float tm = swzmax16(fmaxf(s0[r], s1[r]));
            const float mn = fmaxf(mrow[r], tm);
            alpha[r] = __expf(mrow[r] - mn);
            mrow[r] = mn;
            s0[r] = __expf(s0[r] - mn);
            s1[r] = __expf(s1[r] - mn);
        }

        // ---- P: C-layout -> A-layout via per-wave LDS transpose ----
#pragma unroll
        for (int r = 0; r < 8; ++r) {
            const int row = r + 8 * hi;
            myp[row * 40 + lo16]      = f2bf(s0[r]);
            myp[row * 40 + lo16 + 16] = f2bf(s1[r]);
        }
        Frag16 pf;
        load_frag_bf16(pf, myp + lo16 * 40 + kb);

        // ---- row sums of P via WMMA against ones ----
        f32x8 rs = bwmma(pf, ones, zero8());
#pragma unroll
        for (int r = 0; r < 8; ++r)
            lrow[r] = lrow[r] * alpha[r] + rs[r];

        // ---- O = diag(alpha) O + P V ----
#pragma unroll
        for (int t = 0; t < 4; ++t) {
            Frag16 vf;
            load_frag_bf16(vf, &vbuf[cur][(16 * t + lo16) * 40 + kb]);
#pragma unroll
            for (int r = 0; r < 8; ++r) o[t][r] *= alpha[r];
            o[t] = bwmma(pf, vf, o[t]);
        }

        // publish next tile: my DMA done, then block-wide rendezvous
        wait_async0();
        __syncthreads();
    }

    // ---- epilogue: O /= l, write fp32 [b, s, h*64+dh] ----
#pragma unroll
    for (int r = 0; r < 8; ++r) {
        const float inv = 1.0f / lrow[r];
        const int q = qbase + r + 8 * hi;
        float* orow = out + (size_t)(bb * 2048 + q) * 1024 + hh * 64;
#pragma unroll
        for (int t = 0; t < 4; ++t)
            orow[16 * t + lo16] = o[t][r] * inv;
    }
}

// ---------------------------------------------------------------------------
extern "C" void kernel_launch(void* const* d_in, const int* in_sizes, int n_in,
                              void* d_out, int out_size, void* d_ws, size_t ws_size,
                              hipStream_t stream) {
    (void)in_sizes; (void)n_in; (void)out_size; (void)ws_size;

    const float* Xsrc[3] = { (const float*)d_in[0], (const float*)d_in[1], (const float*)d_in[2] };
    const float* Wsrc[3] = { (const float*)d_in[3], (const float*)d_in[5], (const float*)d_in[7] };
    const float* Bsrc[3] = { (const float*)d_in[4], (const float*)d_in[6], (const float*)d_in[8] };

    // Workspace layout (elements of ushort):
    //   Qp, Kp, Vt : 3 x 8192*1024 bf16 projections (16 MB each)
    //   Xbf        : 8192*1024 bf16 activation staging (reused per projection)
    //   Wbf        : 1024*1024 bf16 weight staging (reused per projection)
    const size_t MD = (size_t)8192 * 1024;
    unsigned short* Qp  = (unsigned short*)d_ws;
    unsigned short* Kp  = Qp + MD;
    unsigned short* Vt  = Kp + MD;
    unsigned short* Xbf = Vt + MD;
    unsigned short* Wbf = Xbf + MD;

    unsigned short* Outp[3] = { Qp, Kp, Vt };
    const int   vtFlag[3]   = { 0, 0, 1 };
    const float oscale[3]   = { 0.125f, 1.0f, 1.0f };   // 1/sqrt(64) folded into Q

    dim3 bGemm(256, 1, 1);
    dim3 gGemm(256, 2, 1);           // M/32 x N/512
    for (int p = 0; p < 3; ++p) {
        cvt_bf16_kernel<<<dim3(4096), dim3(256), 0, stream>>>(Xsrc[p], Xbf);  // 8M elems
        cvt_bf16_kernel<<<dim3(512),  dim3(256), 0, stream>>>(Wsrc[p], Wbf);  // 1M elems
        qkv_proj_kernel<<<gGemm, bGemm, 0, stream>>>(Xbf, Wbf, Bsrc[p],
                                                     Outp[p], vtFlag[p], oscale[p]);
    }

    dim3 gAttn(1024, 1, 1);          // B*H*(S/128)
    dim3 bAttn(256, 1, 1);           // 8 waves
    attn_kernel<<<gAttn, bAttn, 0, stream>>>(Qp, Kp, Vt, (float*)d_out);
}
